// AutoCorrelation_19095424598087
// MI455X (gfx1250) — compile-verified, matrix operand order checked by reading, not run on hardware
//
#include <hip/hip_runtime.h>
#include <hip/hip_bf16.h>

// ---------------------------------------------------------------------------
// AutoCorrelation (Autoformer) for MI455X / gfx1250, wave32 + WMMA bf16.
// FFT correlation replaced by per-batch Gram GEMM  G = Q K^T  (WMMA) followed
// by an anti-diagonal reduction:  mean_value[b,tau] = (1/D) sum_t G[b,t,t-tau].
// Gram GEMM stages tiles through LDS via the Tensor Data Mover (TDM),
// double-buffered, synchronized with s_wait_tensorcnt + workgroup barriers.
// ---------------------------------------------------------------------------

typedef __bf16 bf16_t;
typedef __attribute__((ext_vector_type(16))) __bf16 bf16x16;
typedef __attribute__((ext_vector_type(8)))  __bf16 bf16x8;
typedef __attribute__((ext_vector_type(8)))  float  floatx8;
typedef __attribute__((ext_vector_type(4)))  unsigned int u32x4;
typedef __attribute__((ext_vector_type(8)))  unsigned int u32x8;

#define B_   4
#define L_   4096
#define D_   512
#define TOPK 8

#define KC      32     // k-chunk staged per TDM transfer
#define LDS_ROW 40     // 32 data + 8 pad bf16 per row (TDM pad: 4 dw per 16 dw)

// ---- WMMA fragment loader -------------------------------------------------
// Row-major operand, contraction dim contiguous (stride ld between rows).
// CDNA5 16-bit A/B layout for 16x16x32:
//   lane<16  : row = lane,    K = {0..7, 16..23}
//   lane>=16 : row = lane-16, K = {8..15, 24..31}
__device__ __forceinline__ bf16x16 load_frag_rm(const bf16_t* __restrict__ p, int ld) {
  const int lane  = threadIdx.x & 31;
  const int row   = lane & 15;
  const int khalf = lane >> 4;
  const bf16_t* q = p + (size_t)row * ld + (khalf << 3);
  union { bf16x16 v; bf16x8 h[2]; } u;
  u.h[0] = *(const bf16x8*)(q);
  u.h[1] = *(const bf16x8*)(q + 16);
  return u.v;
}

// ---- TDM: stage a (tile_dim0=KC x tile_dim1=128) bf16 tile into LDS -------
// D# per CDNA5 ISA sec.8. 2D tile, row-major source with
// row stride D_ elements; LDS rows padded 4 DWORDs per 16 DWORDs -> LDS_ROW.
__device__ __forceinline__ void tdm_load_tile(unsigned long long gaddr, unsigned lds_addr) {
  u32x4 g0;
  g0[0] = 1u;                                              // count=1, user desc
  g0[1] = lds_addr;                                        // LDS byte address
  g0[2] = (unsigned)gaddr;                                 // global addr lo
  g0[3] = ((unsigned)(gaddr >> 32) & 0x01FFFFFFu)
        | (2u << 30);                                      // type=2 ("image")
  u32x8 g1;
  g1[0] = (1u << 16)        // data_size = 1 -> 2 bytes
        | (1u << 20)        // pad_enable
        | (3u << 22)        // pad_interval code 3 -> every 16 DWORDs
        | (3u << 25);       // pad_amount  code 3 -> 4 DWORDs (8 bf16)
  g1[1] = ((unsigned)D_ & 0xFFFFu) << 16;                  // tensor_dim0 = 512 (lo16)
  g1[2] = ((unsigned)L_ & 0xFFFFu) << 16;                  // dim0 hi=0 | tensor_dim1 = 4096 (lo16)
  g1[3] = ((unsigned)KC) << 16;                            // dim1 hi=0 | tile_dim0 = KC
  g1[4] = 128u;                                            // tile_dim1 = 128, tile_dim2 = 0
  g1[5] = (unsigned)D_;                                    // tensor_dim0_stride = 512
  g1[6] = 0u;                                              // stride hi | dim1_stride lo (unused, 2D)
  g1[7] = 0u;
  u32x4 g2;                                                // 3rd/4th dims disabled
  g2[0] = 1u; g2[1] = 1u; g2[2] = 0u; g2[3] = 0u;
  u32x4 g3;
  g3[0] = 0u; g3[1] = 0u; g3[2] = 0u; g3[3] = 0u;
  asm volatile("tensor_load_to_lds %0, %1, %2, %3"
               :
               : "s"(g0), "s"(g1), "s"(g2), "s"(g3)
               : "memory");
}

// ---- elementwise convert --------------------------------------------------
__global__ __launch_bounds__(256) void f32_to_bf16_kernel(
    const float* __restrict__ in, bf16_t* __restrict__ out, int n) {
  int i = blockIdx.x * 256 + threadIdx.x;
  if (i < n) out[i] = (bf16_t)in[i];
}

// W (K x N row-major, 512x512) -> WT (N x K row-major, bf16)
__global__ __launch_bounds__(256) void transpose_w_kernel(
    const float* __restrict__ W, bf16_t* __restrict__ WT) {
  int i = blockIdx.x * 256 + threadIdx.x;   // 512*512 threads
  int k = i >> 9;
  int n = i & 511;
  WT[((size_t)n << 9) + k] = (bf16_t)W[i];
}

// ---- generic NT GEMM: C[m,n] = sum_k A[m,k]*B[n,k] + bias[n] --------------
// A: M x K bf16 rm, B: N x K bf16 rm. Block = 256 thr = 8 waves (4x2),
// wave tile 32x64 (2x4 WMMA), block tile 128x128. Operands are L2-resident;
// fragments loaded straight from global.
__global__ __launch_bounds__(256) void gemm_nt_kernel(
    const bf16_t* __restrict__ A, const bf16_t* __restrict__ Bm,
    const float* __restrict__ bias,
    float* __restrict__ Cf, bf16_t* __restrict__ Cb,
    int M, int N, int K)
{
  const int lane = threadIdx.x & 31;
  const int wid  = threadIdx.x >> 5;
  const int wm   = wid >> 1;                 // 0..3 -> 32-row strip
  const int wn   = wid & 1;                  // 0..1 -> 64-col strip
  const int tm   = blockIdx.y * 128 + wm * 32;
  const int tn   = blockIdx.x * 128 + wn * 64;

  const bf16_t* Ap = A  + (size_t)tm * K;
  const bf16_t* Bp = Bm + (size_t)tn * K;

  floatx8 acc[2][4] = {};
  for (int k = 0; k < K; k += 32) {
    bf16x16 a0 = load_frag_rm(Ap + k,                   K);
    bf16x16 a1 = load_frag_rm(Ap + (size_t)16 * K + k,  K);
    bf16x16 b0 = load_frag_rm(Bp + k,                   K);
    bf16x16 b1 = load_frag_rm(Bp + (size_t)16 * K + k,  K);
    bf16x16 b2 = load_frag_rm(Bp + (size_t)32 * K + k,  K);
    bf16x16 b3 = load_frag_rm(Bp + (size_t)48 * K + k,  K);
    acc[0][0] = __builtin_amdgcn_wmma_f32_16x16x32_bf16(false, a0, false, b0, (short)0, acc[0][0], false, false);
    acc[0][1] = __builtin_amdgcn_wmma_f32_16x16x32_bf16(false, a0, false, b1, (short)0, acc[0][1], false, false);
    acc[0][2] = __builtin_amdgcn_wmma_f32_16x16x32_bf16(false, a0, false, b2, (short)0, acc[0][2], false, false);
    acc[0][3] = __builtin_amdgcn_wmma_f32_16x16x32_bf16(false, a0, false, b3, (short)0, acc[0][3], false, false);
    acc[1][0] = __builtin_amdgcn_wmma_f32_16x16x32_bf16(false, a1, false, b0, (short)0, acc[1][0], false, false);
    acc[1][1] = __builtin_amdgcn_wmma_f32_16x16x32_bf16(false, a1, false, b1, (short)0, acc[1][1], false, false);
    acc[1][2] = __builtin_amdgcn_wmma_f32_16x16x32_bf16(false, a1, false, b2, (short)0, acc[1][2], false, false);
    acc[1][3] = __builtin_amdgcn_wmma_f32_16x16x32_bf16(false, a1, false, b3, (short)0, acc[1][3], false, false);
  }

  // C/D layout: vgpr r, lane l -> (M = r + 8*(l>>4), N = l&15) within 16x16 tile
  const int col  = lane & 15;
  const int rofs = (lane >> 4) << 3;
  #pragma unroll
  for (int mi = 0; mi < 2; ++mi) {
    #pragma unroll
    for (int ni = 0; ni < 4; ++ni) {
      int n = tn + ni * 16 + col;
      float bv = bias[n];
      #pragma unroll
      for (int r = 0; r < 8; ++r) {
        int m = tm + mi * 16 + rofs + r;
        float v = acc[mi][ni][r] + bv;
        size_t off = (size_t)m * N + n;
        if (Cf) Cf[off] = v;
        if (Cb) Cb[off] = (bf16_t)v;
      }
    }
  }
}

// ---- Gram GEMM (TDM-staged) + anti-diagonal reduction ---------------------
// Per batch: G[t,t'] = sum_d Q[b,t,d]*K[b,t',d]; accumulate into
// mean_sum[b, (t - t') mod L]. Block tile 128x128 -> 255 distinct diagonals.
__global__ __launch_bounds__(256) void gram_corr_kernel(
    const bf16_t* __restrict__ Qb, const bf16_t* __restrict__ Kb,
    float* __restrict__ mean_sum)
{
  __shared__ bf16_t ldsA[2][128 * LDS_ROW];
  __shared__ bf16_t ldsB[2][128 * LDS_ROW];
  __shared__ float  sdiag[255];

  const int b    = blockIdx.z;
  const int lane = threadIdx.x & 31;
  const int wid  = threadIdx.x >> 5;
  const int wm   = wid >> 1;               // 0..3
  const int wn   = wid & 1;                // 0..1
  const int tm   = blockIdx.y * 128;
  const int tn   = blockIdx.x * 128;

  const unsigned long long baseA =
      (unsigned long long)(uintptr_t)(Qb + (size_t)b * L_ * D_ + (size_t)tm * D_);
  const unsigned long long baseB =
      (unsigned long long)(uintptr_t)(Kb + (size_t)b * L_ * D_ + (size_t)tn * D_);
  const unsigned addrA0 = (unsigned)(uintptr_t)(void*)&ldsA[0][0];
  const unsigned addrA1 = (unsigned)(uintptr_t)(void*)&ldsA[1][0];
  const unsigned addrB0 = (unsigned)(uintptr_t)(void*)&ldsB[0][0];
  const unsigned addrB1 = (unsigned)(uintptr_t)(void*)&ldsB[1][0];

  // prologue: stage chunk 0 into buffer 0 (single wave issues; TDM is per-wave)
  if (wid == 0) {
    tdm_load_tile(baseA, addrA0);
    tdm_load_tile(baseB, addrB0);
  }

  floatx8 acc[2][4] = {};
  const int NCHUNK = D_ / KC;              // 16
  for (int c = 0; c < NCHUNK; ++c) {
    const int buf = c & 1;
    if (wid == 0) {
      if (c + 1 < NCHUNK) {
        unsigned long long go = (unsigned long long)(c + 1) * (KC * 2);
        tdm_load_tile(baseA + go, buf ? addrA0 : addrA1);
        tdm_load_tile(baseB + go, buf ? addrB0 : addrB1);
        __builtin_amdgcn_s_wait_tensorcnt(2);   // chunk c landed; c+1 in flight
      } else {
        __builtin_amdgcn_s_wait_tensorcnt(0);
      }
    }
    __syncthreads();                            // release consumers

    const bf16_t* At = &ldsA[buf][0];
    const bf16_t* Bt = &ldsB[buf][0];
    bf16x16 a0 = load_frag_rm(At + (wm * 32 +  0) * LDS_ROW, LDS_ROW);
    bf16x16 a1 = load_frag_rm(At + (wm * 32 + 16) * LDS_ROW, LDS_ROW);
    bf16x16 b0 = load_frag_rm(Bt + (wn * 64 +  0) * LDS_ROW, LDS_ROW);
    bf16x16 b1 = load_frag_rm(Bt + (wn * 64 + 16) * LDS_ROW, LDS_ROW);
    bf16x16 b2 = load_frag_rm(Bt + (wn * 64 + 32) * LDS_ROW, LDS_ROW);
    bf16x16 b3 = load_frag_rm(Bt + (wn * 64 + 48) * LDS_ROW, LDS_ROW);
    acc[0][0] = __builtin_amdgcn_wmma_f32_16x16x32_bf16(false, a0, false, b0, (short)0, acc[0][0], false, false);
    acc[0][1] = __builtin_amdgcn_wmma_f32_16x16x32_bf16(false, a0, false, b1, (short)0, acc[0][1], false, false);
    acc[0][2] = __builtin_amdgcn_wmma_f32_16x16x32_bf16(false, a0, false, b2, (short)0, acc[0][2], false, false);
    acc[0][3] = __builtin_amdgcn_wmma_f32_16x16x32_bf16(false, a0, false, b3, (short)0, acc[0][3], false, false);
    acc[1][0] = __builtin_amdgcn_wmma_f32_16x16x32_bf16(false, a1, false, b0, (short)0, acc[1][0], false, false);
    acc[1][1] = __builtin_amdgcn_wmma_f32_16x16x32_bf16(false, a1, false, b1, (short)0, acc[1][1], false, false);
    acc[1][2] = __builtin_amdgcn_wmma_f32_16x16x32_bf16(false, a1, false, b2, (short)0, acc[1][2], false, false);
    acc[1][3] = __builtin_amdgcn_wmma_f32_16x16x32_bf16(false, a1, false, b3, (short)0, acc[1][3], false, false);
    __syncthreads();                            // all consumed before next overwrite
  }

  // anti-diagonal reduction within the 128x128 tile
  for (int t = threadIdx.x; t < 255; t += 256) sdiag[t] = 0.f;
  __syncthreads();

  const int col  = lane & 15;
  const int rofs = (lane >> 4) << 3;
  #pragma unroll
  for (int mi = 0; mi < 2; ++mi) {
    #pragma unroll
    for (int ni = 0; ni < 4; ++ni) {
      #pragma unroll
      for (int r = 0; r < 8; ++r) {
        int i = wm * 32 + mi * 16 + rofs + r;   // [0,128)
        int j = wn * 64 + ni * 16 + col;        // [0,128)
        atomicAdd(&sdiag[i - j + 127], acc[mi][ni][r]);
      }
    }
  }
  __syncthreads();

  for (int t = threadIdx.x; t < 255; t += 256) {
    int tau = (tm + t - 127 - tn) & (L_ - 1);   // two's-complement & == mod L
    atomicAdd(&mean_sum[b * L_ + tau], sdiag[t]);
  }
}

// ---- top-8 over batch-mean + per-batch softmax ----------------------------
__global__ __launch_bounds__(256) void topk_softmax_kernel(
    const float* __restrict__ ms, int* __restrict__ idx_out, float* __restrict__ w_out)
{
  __shared__ float tot[L_];
  __shared__ float bval[256];
  __shared__ int   bidx[256];
  __shared__ int   sel[TOPK];
  const int tid = threadIdx.x;

  for (int t = tid; t < L_; t += 256) {
    float s = 0.f;
    for (int b = 0; b < B_; ++b) s += ms[b * L_ + t];
    tot[t] = s;
  }
  __syncthreads();

  for (int r = 0; r < TOPK; ++r) {
    float best = -3.4e38f; int bi = 0;
    for (int t = tid; t < L_; t += 256) {
      float v = tot[t];
      if (v > best) { best = v; bi = t; }
    }
    bval[tid] = best; bidx[tid] = bi;
    __syncthreads();
    if (tid == 0) {
      float bb = bval[0]; int ii = bidx[0];
      for (int j2 = 1; j2 < 256; ++j2)
        if (bval[j2] > bb) { bb = bval[j2]; ii = bidx[j2]; }
      sel[r] = ii;
      tot[ii] = -3.4e38f;
      idx_out[r] = ii;
    }
    __syncthreads();
  }

  if (tid < B_) {
    int b = tid;
    float mv[TOPK];
    float mx = -3.4e38f;
    for (int k2 = 0; k2 < TOPK; ++k2) {
      mv[k2] = ms[b * L_ + sel[k2]] * (1.0f / (float)D_);
      mx = fmaxf(mx, mv[k2]);
    }
    float den = 0.f;
    for (int k2 = 0; k2 < TOPK; ++k2) { mv[k2] = __expf(mv[k2] - mx); den += mv[k2]; }
    float inv = 1.0f / den;
    for (int k2 = 0; k2 < TOPK; ++k2) w_out[b * TOPK + k2] = mv[k2] * inv;
  }
}

// ---- weighted delay-gather of V ------------------------------------------
// agg[b,l,d] = sum_k w[b,k] * V[b,(l+idx[k])%L, d]  -> bf16 for output GEMM
__global__ __launch_bounds__(256) void gather_agg_kernel(
    const float* __restrict__ V, const int* __restrict__ idx,
    const float* __restrict__ w, bf16_t* __restrict__ aggb)
{
  int gid = blockIdx.x * 256 + threadIdx.x;      // over B*L*D = 2^23
  int d = gid & (D_ - 1);
  int l = (gid >> 9) & (L_ - 1);
  int b = gid >> 21;
  float s = 0.f;
  #pragma unroll
  for (int k = 0; k < TOPK; ++k) {
    int lk = (l + idx[k]) & (L_ - 1);
    s += w[b * TOPK + k] * V[((size_t)b * L_ + lk) * D_ + d];
  }
  aggb[gid] = (bf16_t)s;
}

// ---------------------------------------------------------------------------
extern "C" void kernel_launch(void* const* d_in, const int* in_sizes, int n_in,
                              void* d_out, int out_size, void* d_ws, size_t ws_size,
                              hipStream_t stream) {
  const float* q_in = (const float*)d_in[0];
  const float* k_in = (const float*)d_in[1];
  const float* v_in = (const float*)d_in[2];
  const float* Wq = (const float*)d_in[3];
  const float* bq = (const float*)d_in[4];
  const float* Wk = (const float*)d_in[5];
  const float* bk = (const float*)d_in[6];
  const float* Wv = (const float*)d_in[7];
  const float* bv = (const float*)d_in[8];
  const float* Wo = (const float*)d_in[9];
  const float* bo = (const float*)d_in[10];

  const size_t NE = (size_t)B_ * L_ * D_;        // 8388608
  char* ws = (char*)d_ws;
  size_t off = 0;
  auto alloc = [&](size_t bytes) -> void* {
    void* p = ws + off;
    off = (off + bytes + 255) & ~(size_t)255;
    return p;
  };

  bf16_t* Xqb = (bf16_t*)alloc(NE * 2);
  bf16_t* Xkb = (bf16_t*)alloc(NE * 2);
  bf16_t* Xvb = (bf16_t*)alloc(NE * 2);
  bf16_t* WqT = (bf16_t*)alloc((size_t)D_ * D_ * 2);
  bf16_t* WkT = (bf16_t*)alloc((size_t)D_ * D_ * 2);
  bf16_t* WvT = (bf16_t*)alloc((size_t)D_ * D_ * 2);
  bf16_t* WoT = (bf16_t*)alloc((size_t)D_ * D_ * 2);
  bf16_t* Qb  = (bf16_t*)alloc(NE * 2);
  bf16_t* Kb  = (bf16_t*)alloc(NE * 2);
  float*  Vf  = (float*) alloc(NE * 4);
  bf16_t* Agg = (bf16_t*)alloc(NE * 2);
  float*  Ms  = (float*) alloc((size_t)B_ * L_ * 4);
  int*    Idx = (int*)   alloc(256);
  float*  Wsm = (float*) alloc(256);

  const int n_elem = (int)NE;
  const int cvt_blocks = (n_elem + 255) / 256;

  // 1) f32 -> bf16 copies of the three inputs
  f32_to_bf16_kernel<<<cvt_blocks, 256, 0, stream>>>(q_in, Xqb, n_elem);
  f32_to_bf16_kernel<<<cvt_blocks, 256, 0, stream>>>(k_in, Xkb, n_elem);
  f32_to_bf16_kernel<<<cvt_blocks, 256, 0, stream>>>(v_in, Xvb, n_elem);

  // 2) transpose weights to N x K bf16 (all GEMMs become NT)
  const int wt_blocks = (D_ * D_) / 256;
  transpose_w_kernel<<<wt_blocks, 256, 0, stream>>>(Wq, WqT);
  transpose_w_kernel<<<wt_blocks, 256, 0, stream>>>(Wk, WkT);
  transpose_w_kernel<<<wt_blocks, 256, 0, stream>>>(Wv, WvT);
  transpose_w_kernel<<<wt_blocks, 256, 0, stream>>>(Wo, WoT);

  // 3) projections: (B*L x D) @ (D x D)^T(NT) + bias
  dim3 gproj(D_ / 128, (B_ * L_) / 128);
  gemm_nt_kernel<<<gproj, 256, 0, stream>>>(Xqb, WqT, bq, nullptr, Qb, B_ * L_, D_, D_);
  gemm_nt_kernel<<<gproj, 256, 0, stream>>>(Xkb, WkT, bk, nullptr, Kb, B_ * L_, D_, D_);
  gemm_nt_kernel<<<gproj, 256, 0, stream>>>(Xvb, WvT, bv, Vf, nullptr, B_ * L_, D_, D_);

  // 4) Gram correlation: zero accumulator, then per-batch G=QK^T reduced to
  //    mean_sum[b,tau]
  hipMemsetAsync(Ms, 0, (size_t)B_ * L_ * 4, stream);
  dim3 ggram(L_ / 128, L_ / 128, B_);
  gram_corr_kernel<<<ggram, 256, 0, stream>>>(Qb, Kb, Ms);

  // 5) global top-8 delays + per-batch softmax weights
  topk_softmax_kernel<<<1, 256, 0, stream>>>(Ms, Idx, Wsm);

  // 6) weighted delay-gather of V -> bf16
  gather_agg_kernel<<<cvt_blocks, 256, 0, stream>>>(Vf, Idx, Wsm, Agg);

  // 7) output projection -> d_out (f32)
  gemm_nt_kernel<<<gproj, 256, 0, stream>>>(Agg, WoT, bo, (float*)d_out, nullptr, B_ * L_, D_, D_);
}
